// GlobalNet_38199439131028
// MI455X (gfx1250) — compile-verified, matrix operand order checked by reading, not run on hardware
//
#include <hip/hip_runtime.h>
#include <hip/hip_bf16.h>

// ---- CDNA5 WMMA vector types (wave32) ----
typedef __attribute__((ext_vector_type(16))) __bf16 v16bf;
typedef __attribute__((ext_vector_type(2)))  __bf16 bf16x2;
typedef __attribute__((ext_vector_type(8)))  float  v8f;

// Problem constants
#define Bn   16
#define Tn   64
#define Ln   512
#define Cn   64
#define Hn   150
#define G3   450     // 3*H
#define OUTn 24
#define MTOT (Bn*Ln)         // 8192 rows per step
#define KX   (5*Cn)          // 320 : input-conv K  (multiple of 64 -> clean tile split)
#define KH   (5*Hn)          // 750 : hidden-conv K
#define KTOT (KX+KH)         // 1070
#define KPAD 1088            // 17 * 64
#define NTOT 600             // [r,z combined 0..299][x->n 300..449][h->n 450..599]
#define NPAD 640             // 5 * 128

// ---------------------------------------------------------------------------
// Build combined bf16 weight matrix Wg[KPAD][NPAD].
//   rows 0..319   : ki (kw*64+c)      ; rows 320..1069 : kh (kw*150+hc)
//   cols 0..299   : r,z gates (both ki and kh contribute)
//   cols 300..449 : n gate, input part only  (ki cols 300..449)
//   cols 450..599 : n gate, hidden part only (kh cols 300..449)
// ---------------------------------------------------------------------------
__global__ void prep_weights(const float* __restrict__ ki,
                             const float* __restrict__ kh,
                             __bf16* __restrict__ Wg) {
    int idx = blockIdx.x * 256 + threadIdx.x;
    if (idx >= KPAD * NPAD) return;
    int k = idx / NPAD;
    int g = idx - k * NPAD;
    float v = 0.f;
    if (g < 300) {
        if (k < KX) {
            int kw = k >> 6, c = k & 63;
            v = ki[(kw * Cn + c) * G3 + g];
        } else if (k < KTOT) {
            int k2 = k - KX, kw = k2 / Hn, hc = k2 - kw * Hn;
            v = kh[(kw * Hn + hc) * G3 + g];
        }
    } else if (g < 450) {                 // input -> n  (iin)
        if (k < KX) {
            int kw = k >> 6, c = k & 63;
            v = ki[(kw * Cn + c) * G3 + g];           // g == 300..449 matches ki col
        }
    } else if (g < NTOT) {                // hidden -> n (hin)
        if (k >= KX && k < KTOT) {
            int k2 = k - KX, kw = k2 / Hn, hc = k2 - kw * Hn;
            v = kh[(kw * Hn + hc) * G3 + (g - 150)];  // kh col 300..449
        }
    }
    Wg[idx] = (__bf16)v;
}

__global__ void zero_f32(float* __restrict__ p, int n) {
    int i = blockIdx.x * 256 + threadIdx.x;
    if (i < n) p[i] = 0.f;
}

// ---------------------------------------------------------------------------
// One recurrent step as a fused GEMM:
//   gates[8192 x 600] = A[8192 x 1070] * Wg[1070 x 600]
// A row m = (b,l):  k<320 -> xs[b,t,l+kw-2,c] (SAME pad),  else h[b,l+kw-2,hc].
// WG tile 64(M) x 128(N); 8 waves = 4(M) x 2(N); each wave: 16x64 strip,
// 4 v8f accumulators, 8 x v_wmma_f32_16x16x32_bf16 per barrier pair (K step 64).
// K=320 region boundary is 64-aligned -> every staged K-tile is purely-x or
// purely-h (tile-uniform branch, no per-lane region divergence).
// ---------------------------------------------------------------------------
__global__ __launch_bounds__(256) void step_gemm(
    const float* __restrict__ xs, const float* __restrict__ h,
    const __bf16* __restrict__ Wg, float* __restrict__ gates, int t) {

    __shared__ __bf16 As[64 * 64];    // [row][k]   8 KB
    __shared__ __bf16 Bs[128 * 64];   // [n][k]    16 KB (K-transposed for pair reads)

    const int tid  = threadIdx.x;
    const int lane = tid & 31;
    const int wave = tid >> 5;
    const int mi   = wave >> 1;          // 0..3 : M sub-tile (16 rows each)
    const int ni   = wave & 1;           // 0..1 : N strip (64 cols each)
    const int half = lane >> 4;          // 0/1
    const int mr   = lane & 15;          // row (A) / col (B,C) within tile
    const int mBase = blockIdx.x * 64;
    const int nBase = blockIdx.y * 128;

    v8f acc[4] = {v8f{}, v8f{}, v8f{}, v8f{}};

    for (int kBase = 0; kBase < KPAD; kBase += 64) {
        // ---- stage A tile (64 rows x 64 K): region is uniform per K-tile
        if (kBase < KX) {
            // pure input-conv region
            for (int i = tid; i < 64 * 64; i += 256) {
                int r = i >> 6, kk = i & 63;
                int row = mBase + r;
                int b = row >> 9, l = row & 511;
                int k = kBase + kk;
                int kw = k >> 6, c = k & 63;
                int lp = l + kw - 2;
                float v = 0.f;
                if (lp >= 0 && lp < Ln)
                    v = xs[((((size_t)b * Tn + t) * Ln + lp) << 6) + c];
                As[i] = (__bf16)v;
            }
        } else {
            // pure hidden-conv region (tail k2 >= 750 zero-padded)
            for (int i = tid; i < 64 * 64; i += 256) {
                int r = i >> 6, kk = i & 63;
                int row = mBase + r;
                int b = row >> 9, l = row & 511;
                int k2 = kBase - KX + kk;
                float v = 0.f;
                if (k2 < KH) {
                    int kw = k2 / Hn, hc = k2 - kw * Hn;
                    int lp = l + kw - 2;
                    if (lp >= 0 && lp < Ln)
                        v = h[(size_t)(b * Ln + lp) * Hn + hc];
                }
                As[i] = (__bf16)v;
            }
        }
        // ---- stage B tile (64 K x 128 N) transposed -> Bs[n][k]
        for (int i = tid; i < 128 * 64; i += 256) {
            int kk = i >> 7, nn = i & 127;       // consecutive tid -> contiguous Wg
            Bs[nn * 64 + kk] = Wg[(size_t)(kBase + kk) * NPAD + nBase + nn];
        }
        __syncthreads();

        // prefetch next weight K-tile into cache (global_prefetch_b8)
        if (kBase + 64 < KPAD)
            __builtin_prefetch(&Wg[(size_t)(kBase + 64) * NPAD + nBase + (tid & 127)], 0, 0);

        // ---- 2 K-substeps x 4 N-tiles = 8 WMMAs per barrier pair
        #pragma unroll
        for (int ks = 0; ks < 64; ks += 32) {
            // A fragment per ISA 16-bit 16x32 layout:
            // VGPR v holds K pair base = 2v (+8 for v>=4) + 8*half
            v16bf afrag;
            #pragma unroll
            for (int v = 0; v < 8; ++v) {
                int kk = ks + v * 2 + (v >= 4 ? 8 : 0) + half * 8;
                bf16x2 pa = *(const bf16x2*)&As[(mi * 16 + mr) * 64 + kk];
                afrag[2 * v] = pa.x; afrag[2 * v + 1] = pa.y;
            }
            #pragma unroll
            for (int j = 0; j < 4; ++j) {
                v16bf bfrag;
                #pragma unroll
                for (int v = 0; v < 8; ++v) {
                    int kk = ks + v * 2 + (v >= 4 ? 8 : 0) + half * 8;
                    bf16x2 pb = *(const bf16x2*)&Bs[(ni * 64 + j * 16 + mr) * 64 + kk];
                    bfrag[2 * v] = pb.x; bfrag[2 * v + 1] = pb.y;
                }
                acc[j] = __builtin_amdgcn_wmma_f32_16x16x32_bf16(
                    /*neg_a=*/false, afrag, /*neg_b=*/false, bfrag,
                    /*c_mod=*/(short)0, acc[j], /*reuse_a=*/false, /*reuse_b=*/false);
            }
        }
        __syncthreads();
    }

    // ---- store C: VGPR r -> M = r + 8*half, N = mr
    #pragma unroll
    for (int j = 0; j < 4; ++j) {
        #pragma unroll
        for (int r = 0; r < 8; ++r) {
            int m = mBase + mi * 16 + half * 8 + r;
            int n = nBase + ni * 64 + j * 16 + mr;
            if (n < NTOT) gates[(size_t)m * NTOT + n] = acc[j][r];
        }
    }
}

// ---------------------------------------------------------------------------
// Elementwise gates: r=sig(P0), z=sig(P1), n=tanh(P2 + r*P3); h' = (1-z)n + z h
// ---------------------------------------------------------------------------
__global__ void gate_kernel(const float* __restrict__ P,
                            const float* __restrict__ h_old,
                            float* __restrict__ h_new) {
    int idx = blockIdx.x * 256 + threadIdx.x;
    if (idx >= MTOT * Hn) return;
    int m = idx / Hn, hc = idx - m * Hn;
    const float* p = P + (size_t)m * NTOT;
    float r = 1.f / (1.f + __expf(-p[hc]));
    float z = 1.f / (1.f + __expf(-p[150 + hc]));
    float n = tanhf(p[300 + hc] + r * p[450 + hc]);
    h_new[idx] = (1.f - z) * n + z * h_old[idx];
}

// ---- MLP head (tiny: ~0.4 GFLOP, h/hdn are L2-resident) ----
__global__ void mlp1(const float* __restrict__ h, const float* __restrict__ W1,
                     const float* __restrict__ b1, float* __restrict__ hdn) {
    int idx = blockIdx.x * 256 + threadIdx.x;
    if (idx >= MTOT * Hn) return;
    int m = idx / Hn, j = idx - m * Hn;
    float s = b1[j];
    const float* hr = h + (size_t)m * Hn;
    for (int i = 0; i < Hn; ++i) s += hr[i] * W1[i * Hn + j];
    hdn[idx] = s * (1.f / (1.f + __expf(-s)));   // silu
}

__global__ void mlp2(const float* __restrict__ hdn, const float* __restrict__ W2,
                     const float* __restrict__ b2, float* __restrict__ out) {
    int idx = blockIdx.x * 256 + threadIdx.x;
    if (idx >= MTOT * OUTn) return;
    int m = idx / OUTn, j = idx - m * OUTn;
    float s = b2[j];
    const float* hr = hdn + (size_t)m * Hn;
    for (int i = 0; i < Hn; ++i) s += hr[i] * W2[i * OUTn + j];
    out[idx] = s;    // [B, L, 24, 1] flat == [m][j]
}

// ---------------------------------------------------------------------------
extern "C" void kernel_launch(void* const* d_in, const int* in_sizes, int n_in,
                              void* d_out, int out_size, void* d_ws, size_t ws_size,
                              hipStream_t stream) {
    (void)in_sizes; (void)n_in; (void)out_size; (void)ws_size;
    const float* xs = (const float*)d_in[0];
    const float* ki = (const float*)d_in[1];
    const float* kh = (const float*)d_in[2];
    const float* W1 = (const float*)d_in[3];
    const float* b1 = (const float*)d_in[4];
    const float* W2 = (const float*)d_in[5];
    const float* b2 = (const float*)d_in[6];
    float* out = (float*)d_out;

    char* ws = (char*)d_ws;
    size_t off = 0;
    __bf16* Wg   = (__bf16*)(ws + off); off += (size_t)KPAD * NPAD * 2;   // 1.4 MB
    float* hA    = (float*)(ws + off);  off += (size_t)MTOT * Hn * 4;     // 4.9 MB
    float* hB    = (float*)(ws + off);  off += (size_t)MTOT * Hn * 4;     // 4.9 MB
    float* gates = (float*)(ws + off);  off += (size_t)MTOT * NTOT * 4;   // 19.7 MB
    float* hdn   = (float*)(ws + off);  off += (size_t)MTOT * Hn * 4;     // 4.9 MB

    prep_weights<<<(KPAD * NPAD + 255) / 256, 256, 0, stream>>>(ki, kh, Wg);
    zero_f32<<<(MTOT * Hn + 255) / 256, 256, 0, stream>>>(hA, MTOT * Hn);

    float* cur = hA;
    float* nxt = hB;
    for (int t = 0; t < Tn; ++t) {
        step_gemm<<<dim3(MTOT / 64, NPAD / 128), 256, 0, stream>>>(xs, cur, Wg, gates, t);
        gate_kernel<<<(MTOT * Hn + 255) / 256, 256, 0, stream>>>(gates, cur, nxt);
        float* tmp = cur; cur = nxt; nxt = tmp;
    }
    // T=64 steps -> final state lands back in hA (== cur after even # of swaps)
    mlp1<<<(MTOT * Hn + 255) / 256, 256, 0, stream>>>(cur, W1, b1, hdn);
    mlp2<<<(MTOT * OUTn + 255) / 256, 256, 0, stream>>>(hdn, W2, b2, out);
}
